// SRNN_custom_2284922601665
// MI455X (gfx1250) — matrix operand order repaired and checked by reading, not compiled, not run on hardware
//
#include <hip/hip_runtime.h>

// ---------------- types ----------------
typedef __bf16 v16bf __attribute__((ext_vector_type(16)));
typedef float  v8f   __attribute__((ext_vector_type(8)));

#define HDIM   128
#define BATCH  256
#define TSTEPS 250
#define IDIM   700
#define ODIM   20
#define KC_IN  22   // ceil(700/32)
#define KC_H   4    // 128/32

static __device__ __forceinline__ v8f wmma_bf16(v16bf a, v16bf b, v8f c) {
  // D = A(16x32 bf16) * B(32x16 bf16) + C(16x16 f32)
  return __builtin_amdgcn_wmma_f32_16x16x32_bf16(false, a, false, b, (short)0, c, false, false);
}

// Build a 16x32 bf16 A-fragment from a row-major [16 x HDIM] f32 LDS buffer.
// A layout (16-bit): lane<16 -> M=lane, K = base+{0..7,16..23};
//                    lane>=16 -> M=lane-16, K = base+{8..15,24..31}
static __device__ __forceinline__ v16bf build_afrag(const float* s, int lane, int kc) {
  int m  = lane & 15;
  int kh = (lane < 16) ? 0 : 8;
  const float* p = s + m * HDIM + kc * 32 + kh;
  v16bf a;
#pragma unroll
  for (int j = 0; j < 8; ++j) a[j]     = (__bf16)p[j];
#pragma unroll
  for (int j = 0; j < 8; ++j) a[8 + j] = (__bf16)p[16 + j];
  return a;
}

// ---------------- kernel 0: pack f32 weights -> bf16 B-fragment layout ----------------
// Packed index: ((ntile*kchunks + kc)*32 + lane)*16 + j
// B element for lane L, slot j: N = ntile*16 + (L%16), K = kc*32 + (L<16?0:8) + (j<8 ? j : 8+j)
__global__ __launch_bounds__(256) void pack_wb(const float* __restrict__ W, __bf16* __restrict__ out,
                                               int N, int K, int ntiles, int kchunks) {
  int tid = blockIdx.x * blockDim.x + threadIdx.x;
  int total = ntiles * kchunks * 32 * 16;
  if (tid >= total) return;
  int j    = tid & 15;
  int lane = (tid >> 4) & 31;
  int kc   = (tid >> 9) % kchunks;
  int nt   = (tid >> 9) / kchunks;
  int n = nt * 16 + (lane & 15);
  int k = kc * 32 + ((lane < 16) ? 0 : 8) + ((j < 8) ? j : (8 + j));
  float v = (n < N && k < K) ? W[(size_t)n * K + k] : 0.0f;
  out[tid] = (__bf16)v;
}

// ---------------- kernel 1: IN1 = x @ W_ih1.T + b_ih1 + b_h1h1 ----------------
// One wave per block computes a 16(batch)x128 tile for one (batch-tile, t).
// Output stored in C-fragment layout: ((((bt*T)+t)*8 + nt)*32 + lane)*8 floats.
__global__ __launch_bounds__(32) void in1_gemm(const float* __restrict__ x,
                                               const __bf16* __restrict__ Wp,
                                               const float* __restrict__ b_ih1,
                                               const float* __restrict__ b_h1h1,
                                               float* __restrict__ IN1) {
  int t = blockIdx.x, bt = blockIdx.y, lane = threadIdx.x;
  int n = lane & 15;
  const float* xr = x + ((size_t)(bt * 16 + n) * TSTEPS + t) * IDIM;  // A row for M = lane%16
  int kh = (lane < 16) ? 0 : 8;

  v8f c[8];
#pragma unroll
  for (int nt = 0; nt < 8; ++nt) {
    float bias = b_ih1[nt * 16 + n] + b_h1h1[nt * 16 + n];
#pragma unroll
    for (int r = 0; r < 8; ++r) c[nt][r] = bias;
  }

  // main K chunks (no bounds checks): 0..20 cover K=0..671
  for (int kc = 0; kc < 21; ++kc) {
    const float* p0 = xr + kc * 32 + kh;
    const float* p1 = p0 + 16;
    v16bf a;
#pragma unroll
    for (int j = 0; j < 8; ++j) a[j]     = (__bf16)p0[j];
#pragma unroll
    for (int j = 0; j < 8; ++j) a[8 + j] = (__bf16)p1[j];
#pragma unroll
    for (int nt = 0; nt < 8; ++nt) {
      v16bf w = *(const v16bf*)(Wp + ((size_t)(nt * KC_IN + kc) * 32 + lane) * 16);
      c[nt] = wmma_bf16(a, w, c[nt]);
    }
  }
  // tail chunk kc=21 covers K=672..703, valid up to 699 (zero-pad rest)
  {
    const int kc = 21;
    v16bf a;
#pragma unroll
    for (int j = 0; j < 8; ++j) {
      int k1 = kc * 32 + kh + j;
      int k2 = kc * 32 + 16 + kh + j;
      a[j]     = (__bf16)((k1 < IDIM) ? xr[k1] : 0.0f);
      a[8 + j] = (__bf16)((k2 < IDIM) ? xr[k2] : 0.0f);
    }
#pragma unroll
    for (int nt = 0; nt < 8; ++nt) {
      v16bf w = *(const v16bf*)(Wp + ((size_t)(nt * KC_IN + kc) * 32 + lane) * 16);
      c[nt] = wmma_bf16(a, w, c[nt]);
    }
  }

#pragma unroll
  for (int nt = 0; nt < 8; ++nt)
    *(v8f*)(IN1 + ((((size_t)bt * TSTEPS + t) * 8 + nt) * 32 + lane) * 8) = c[nt];
}

// ---------------- kernel 2: sequential adaptive-LIF scan ----------------
// 16 blocks x 8 waves (256 threads). Block bt owns batch rows bt*16..bt*16+15;
// wave w owns hidden ntile w (16 columns). Weight fragments live in registers.
__global__ __launch_bounds__(256) void srnn_scan(
    const float* __restrict__ IN1,
    const __bf16* __restrict__ W11p, const __bf16* __restrict__ W12p,
    const __bf16* __restrict__ W22p, const __bf16* __restrict__ W2op,
    const float* __restrict__ b_h1h2, const float* __restrict__ b_h2h2,
    const float* __restrict__ b_h2o,
    const float* __restrict__ tau_adp_h1, const float* __restrict__ tau_adp_h2,
    const float* __restrict__ tau_m_h1, const float* __restrict__ tau_m_h2,
    const float* __restrict__ tau_m_o,
    const float* __restrict__ hid1_mem0, const float* __restrict__ hid2_mem0,
    const float* __restrict__ out_mem0,
    float* __restrict__ out) {
  __shared__ float h1m[16 * HDIM], b1l[16 * HDIM], s1[16 * HDIM];
  __shared__ float h2m[16 * HDIM], b2l[16 * HDIM], s2[16 * HDIM];
  __shared__ float omv[16 * 32], accv[16 * ODIM];
  __shared__ float a1[HDIM], r1[HDIM], a2[HDIM], r2[HDIM], bias2[HDIM];
  __shared__ float ao[ODIM], biaso[ODIM];

  int bt   = blockIdx.x;
  int tid  = threadIdx.x;
  int wv   = tid >> 5;      // wave id == owned ntile
  int lane = tid & 31;
  int n    = lane & 15;

  // ---- init state (cooperative, 256 threads) ----
  for (int i = tid; i < 16 * HDIM; i += 256) {
    int m = i >> 7, h = i & 127;
    int row = bt * 16 + m;
    h1m[i] = hid1_mem0[row * HDIM + h];
    h2m[i] = hid2_mem0[row * HDIM + h];
    b1l[i] = 0.01f; b2l[i] = 0.01f;   // b_j0
    s1[i]  = 0.0f;  s2[i]  = 0.0f;
  }
  for (int h = tid; h < HDIM; h += 256) {
    a1[h] = __expf(-1.0f / tau_m_h1[h]);
    r1[h] = __expf(-1.0f / tau_adp_h1[h]);
    a2[h] = __expf(-1.0f / tau_m_h2[h]);
    r2[h] = __expf(-1.0f / tau_adp_h2[h]);
    bias2[h] = b_h1h2[h] + b_h2h2[h];
  }
  for (int o = tid; o < ODIM; o += 256) {
    ao[o] = __expf(-1.0f / tau_m_o[o]);
    biaso[o] = b_h2o[o];
  }
  for (int i = tid; i < 16 * 32; i += 256) omv[i] = 0.0f;
  for (int i = tid; i < 16 * ODIM; i += 256) {
    int m = i / ODIM, o = i % ODIM;
    omv[m * 32 + o] = out_mem0[(bt * 16 + m) * ODIM + o];
    accv[i] = 0.0f;
  }

  // ---- hoist this wave's weight B-fragments into registers ----
  v16bf w11f[KC_H], w12f[KC_H], w22f[KC_H];
#pragma unroll
  for (int kc = 0; kc < KC_H; ++kc) {
    w11f[kc] = *(const v16bf*)(W11p + ((size_t)(wv * KC_H + kc) * 32 + lane) * 16);
    w12f[kc] = *(const v16bf*)(W12p + ((size_t)(wv * KC_H + kc) * 32 + lane) * 16);
    w22f[kc] = *(const v16bf*)(W22p + ((size_t)(wv * KC_H + kc) * 32 + lane) * 16);
  }
  int h = wv * 16 + n;       // this wave's hidden column (per lane)
  __syncthreads();

  for (int t = 0; t < TSTEPS; ++t) {
    // L2 prefetch of next timestep's IN1 tile for this wave
    if (t + 1 < TSTEPS)
      __builtin_prefetch(IN1 + ((((size_t)bt * TSTEPS + (t + 1)) * 8 + wv) * 32 + lane) * 8, 0, 0);

    // (1) spike A-fragments from previous step (all waves need full K)
    v16bf s1f[KC_H], s2f[KC_H];
#pragma unroll
    for (int kc = 0; kc < KC_H; ++kc) {
      s1f[kc] = build_afrag(s1, lane, kc);
      s2f[kc] = build_afrag(s2, lane, kc);
    }
    __syncthreads();  // A: reads done before layer1/2 overwrite s1/s2

    // (2) layer 1, ntile = wv: in1 = IN1(t) + h1s_prev @ W_h1h1.T ; adaptive LIF
    {
      v8f c = *(const v8f*)(IN1 + ((((size_t)bt * TSTEPS + t) * 8 + wv) * 32 + lane) * 8);
#pragma unroll
      for (int kc = 0; kc < KC_H; ++kc) c = wmma_bf16(s1f[kc], w11f[kc], c);
      float A = a1[h], R = r1[h];
#pragma unroll
      for (int r = 0; r < 8; ++r) {
        int m = (lane < 16) ? r : (r + 8);
        int idx = m * HDIM + h;
        float sold = s1[idx], bold = b1l[idx], mold = h1m[idx];
        float bnew = R * bold + (1.0f - R) * sold;
        float Bth  = 0.01f + 1.8f * bnew;                       // b_j0 + beta*b
        float mnew = mold * A + (1.0f - A) * c[r] - Bth * sold; // R_m = 1
        float snew = (mnew - Bth) > 0.0f ? 1.0f : 0.0f;
        b1l[idx] = bnew; h1m[idx] = mnew; s1[idx] = snew;
      }
    }
    __syncthreads();  // B: new s1 visible

    // (3) rebuild layer-1 spike fragments (new h1s)
#pragma unroll
    for (int kc = 0; kc < KC_H; ++kc) s1f[kc] = build_afrag(s1, lane, kc);

    // (4) layer 2, ntile = wv: in2 = h1s @ W_h1h2.T + h2s_prev @ W_h2h2.T ; adaptive LIF
    {
      float bias = bias2[h];
      v8f c;
#pragma unroll
      for (int r = 0; r < 8; ++r) c[r] = bias;
#pragma unroll
      for (int kc = 0; kc < KC_H; ++kc) {
        c = wmma_bf16(s1f[kc], w12f[kc], c);
        c = wmma_bf16(s2f[kc], w22f[kc], c);
      }
      float A = a2[h], R = r2[h];
#pragma unroll
      for (int r = 0; r < 8; ++r) {
        int m = (lane < 16) ? r : (r + 8);
        int idx = m * HDIM + h;
        float sold = s2[idx], bold = b2l[idx], mold = h2m[idx];
        float bnew = R * bold + (1.0f - R) * sold;
        float Bth  = 0.01f + 1.8f * bnew;
        float mnew = mold * A + (1.0f - A) * c[r] - Bth * sold;
        float snew = (mnew - Bth) > 0.0f ? 1.0f : 0.0f;
        b2l[idx] = bnew; h2m[idx] = mnew; s2[idx] = snew;
      }
    }
    __syncthreads();  // C: new s2 visible

    // (5) readout (waves 0,1 only; wave-uniform branch keeps EXEC full for WMMA)
    if (wv < 2) {
      v16bf s2fn[KC_H];
#pragma unroll
      for (int kc = 0; kc < KC_H; ++kc) s2fn[kc] = build_afrag(s2, lane, kc);
      int o = wv * 16 + n;
      bool valid = (o < ODIM);
      float bias = valid ? biaso[o] : 0.0f;
      v8f c;
#pragma unroll
      for (int r = 0; r < 8; ++r) c[r] = bias;
#pragma unroll
      for (int kc = 0; kc < KC_H; ++kc) {
        v16bf w = *(const v16bf*)(W2op + ((size_t)(wv * KC_H + kc) * 32 + lane) * 16);
        c = wmma_bf16(s2fn[kc], w, c);
      }
      if (valid) {
        float A = ao[o];
#pragma unroll
        for (int r = 0; r < 8; ++r) {
          int m = (lane < 16) ? r : (r + 8);
          int idx = m * 32 + o;
          omv[idx] = omv[idx] * A + (1.0f - A) * c[r];
        }
      }
    }
    __syncthreads();  // D: omv visible to softmax

    // (6) softmax accumulate for t > 10 (wave 0, lanes 0..15: one row each)
    if (t > 10 && tid < 16) {
      int m = tid;
      float mx = -3.4e38f;
#pragma unroll
      for (int o = 0; o < ODIM; ++o) mx = fmaxf(mx, omv[m * 32 + o]);
      float e[ODIM];
      float s = 0.0f;
#pragma unroll
      for (int o = 0; o < ODIM; ++o) { e[o] = __expf(omv[m * 32 + o] - mx); s += e[o]; }
      float inv = 1.0f / s;
#pragma unroll
      for (int o = 0; o < ODIM; ++o) accv[m * ODIM + o] += e[o] * inv;
    }
    // next iteration's barriers fence softmax reads from future omv writes
  }

  __syncthreads();
  // ---- write accumulated softmax ----
  for (int i = tid; i < 16 * ODIM; i += 256) {
    int m = i / ODIM, o = i % ODIM;
    out[(bt * 16 + m) * ODIM + o] = accv[i];
  }
}

// ---------------- kernel 3: A_norm = sum|W_h1h1| + sum|W_h2h2| ----------------
__global__ __launch_bounds__(256) void abs_norm(const float* __restrict__ W11,
                                                const float* __restrict__ W22,
                                                float* __restrict__ out) {
  __shared__ float red[256];
  float s = 0.0f;
  for (int i = threadIdx.x; i < HDIM * HDIM; i += 256)
    s += fabsf(W11[i]) + fabsf(W22[i]);
  red[threadIdx.x] = s;
  __syncthreads();
  for (int k = 128; k > 0; k >>= 1) {
    if (threadIdx.x < k) red[threadIdx.x] += red[threadIdx.x + k];
    __syncthreads();
  }
  if (threadIdx.x == 0) out[BATCH * ODIM] = red[0];
}

// ---------------- host launcher ----------------
extern "C" void kernel_launch(void* const* d_in, const int* in_sizes, int n_in,
                              void* d_out, int out_size, void* d_ws, size_t ws_size,
                              hipStream_t stream) {
  (void)in_sizes; (void)n_in; (void)out_size; (void)ws_size;

  const float* x          = (const float*)d_in[0];
  // d_in[1], d_in[2]: A1_mask / A2_mask (all-ones, unused by reference math)
  const float* W_ih1      = (const float*)d_in[3];
  const float* b_ih1      = (const float*)d_in[4];
  const float* W_h1h1     = (const float*)d_in[5];
  const float* b_h1h1     = (const float*)d_in[6];
  const float* W_h1h2     = (const float*)d_in[7];
  const float* b_h1h2     = (const float*)d_in[8];
  const float* W_h2h2     = (const float*)d_in[9];
  const float* b_h2h2     = (const float*)d_in[10];
  const float* W_h2o      = (const float*)d_in[11];
  const float* b_h2o      = (const float*)d_in[12];
  const float* tau_adp_h1 = (const float*)d_in[13];
  const float* tau_adp_h2 = (const float*)d_in[14];
  const float* tau_m_h1   = (const float*)d_in[15];
  const float* tau_m_h2   = (const float*)d_in[16];
  const float* tau_m_o    = (const float*)d_in[17];
  const float* hid1_mem0  = (const float*)d_in[18];
  const float* hid2_mem0  = (const float*)d_in[19];
  const float* out_mem0   = (const float*)d_in[20];

  // workspace layout (bytes)
  char* ws = (char*)d_ws;
  float*  IN1   = (float*)(ws + 0);                        // 64000*128*4 = 32,768,000
  __bf16* Wih1p = (__bf16*)(ws + 32768000);                // 8*22*512*2  =    180,224
  __bf16* W11p  = (__bf16*)(ws + 32948224);                // 8*4*512*2   =     32,768
  __bf16* W12p  = (__bf16*)(ws + 32980992);
  __bf16* W22p  = (__bf16*)(ws + 33013760);
  __bf16* W2op  = (__bf16*)(ws + 33046528);                // 2*4*512*2   =      8,192

  // 1) pack weights to bf16 fragment layout
  pack_wb<<<(8 * KC_IN * 512 + 255) / 256, 256, 0, stream>>>(W_ih1, Wih1p, HDIM, IDIM, 8, KC_IN);
  pack_wb<<<(8 * KC_H  * 512 + 255) / 256, 256, 0, stream>>>(W_h1h1, W11p, HDIM, HDIM, 8, KC_H);
  pack_wb<<<(8 * KC_H  * 512 + 255) / 256, 256, 0, stream>>>(W_h1h2, W12p, HDIM, HDIM, 8, KC_H);
  pack_wb<<<(8 * KC_H  * 512 + 255) / 256, 256, 0, stream>>>(W_h2h2, W22p, HDIM, HDIM, 8, KC_H);
  pack_wb<<<(2 * KC_H  * 512 + 255) / 256, 256, 0, stream>>>(W_h2o,  W2op, ODIM, HDIM, 2, KC_H);

  // 2) bulk input GEMM (time-parallel, HBM-bound on x)
  dim3 g1(TSTEPS, BATCH / 16);
  in1_gemm<<<g1, 32, 0, stream>>>(x, Wih1p, b_ih1, b_h1h1, IN1);

  // 3) sequential recurrent scan (batch x N parallel: 16 WGs x 8 waves)
  srnn_scan<<<BATCH / 16, 256, 0, stream>>>(IN1, W11p, W12p, W22p, W2op,
                                            b_h1h2, b_h2h2, b_h2o,
                                            tau_adp_h1, tau_adp_h2,
                                            tau_m_h1, tau_m_h2, tau_m_o,
                                            hid1_mem0, hid2_mem0, out_mem0,
                                            (float*)d_out);

  // 4) scalar weight norm -> d_out[5120]
  abs_norm<<<1, 256, 0, stream>>>(W_h1h1, W_h2h2, (float*)d_out);
}
